// HyperbolicTransformerLayer_74904229642743
// MI455X (gfx1250) — matrix operand order repaired and checked by reading, not compiled, not run on hardware
//
#include <hip/hip_runtime.h>
#include <math.h>

// ---------------------------------------------------------------------------
// CDNA5 WMMA types (wave32, v_wmma_f32_16x16x32_bf16)
// ---------------------------------------------------------------------------
typedef __attribute__((ext_vector_type(16))) __bf16 v16bf;
typedef __attribute__((ext_vector_type(8)))  __bf16 v8bf;
typedef __attribute__((ext_vector_type(8)))  float  v8f;

static __device__ __forceinline__ __bf16 f2bf(float f) {
  unsigned u = __builtin_bit_cast(unsigned, f);
  unsigned r = (u + 0x7FFFu + ((u >> 16) & 1u)) >> 16;  // round-nearest-even
  return __builtin_bit_cast(__bf16, (unsigned short)r);
}

static __device__ __forceinline__ float artanhf_(float z) {
  return 0.5f * logf((1.0f + z) / (1.0f - z));
}

// Block-wide reductions (wave32). sm must hold >= blockDim/32 floats.
static __device__ __forceinline__ float block_reduce_sum(float v, float* sm) {
  #pragma unroll
  for (int o = 16; o > 0; o >>= 1) v += __shfl_down(v, o, 32);
  int wid = threadIdx.x >> 5, lid = threadIdx.x & 31;
  __syncthreads();
  if (lid == 0) sm[wid] = v;
  __syncthreads();
  float tot = 0.f;
  int nw = (blockDim.x + 31) >> 5;
  for (int i = 0; i < nw; ++i) tot += sm[i];
  return tot;
}

static __device__ __forceinline__ float block_reduce_max(float v, float* sm) {
  #pragma unroll
  for (int o = 16; o > 0; o >>= 1) v = fmaxf(v, __shfl_down(v, o, 32));
  int wid = threadIdx.x >> 5, lid = threadIdx.x & 31;
  __syncthreads();
  if (lid == 0) sm[wid] = v;
  __syncthreads();
  float tot = -3.0e38f;
  int nw = (blockDim.x + 31) >> 5;
  for (int i = 0; i < nw; ++i) tot = fmaxf(tot, sm[i]);
  return tot;
}

// ---------------------------------------------------------------------------
// Pure-bf16 fragment loads: no conversion in the GEMM inner loop.
// A frag (16x32): lane m=lane%16, half=lane/16; base k = half*8;
//   elems j0..3 = k base+0..7 (16B), j4..7 = k base+16..23 (16B).
// B^T frag (32x16, B row-major (N,K)): lane n=lane%16, half=lane/16;
//   base k = half*8; elems = k base+0..7 (16B), k base+16..23 (16B).
// ---------------------------------------------------------------------------
static __device__ __forceinline__ v16bf load_afrag(const __bf16* p) {
  v8bf lo = *(const v8bf*)(p);
  v8bf hi = *(const v8bf*)(p + 16);
  return __builtin_shufflevector(lo, hi, 0, 1, 2, 3, 4, 5, 6, 7,
                                 8, 9, 10, 11, 12, 13, 14, 15);
}

static __device__ __forceinline__ v16bf load_bfrag(const __bf16* p) {
  v8bf lo = *(const v8bf*)(p);
  v8bf hi = *(const v8bf*)(p + 16);
  return __builtin_shufflevector(lo, hi, 0, 1, 2, 3, 4, 5, 6, 7,
                                 8, 9, 10, 11, 12, 13, 14, 15);
}

// ---------------------------------------------------------------------------
// Batched GEMM  C = A * B^T  (bf16 in, f32 out). Operands padded: no K/M
// guards. Each wave owns 16x32 of C (2 accumulators reusing the A fragment);
// a block of W waves covers 16 x (W*32). Wave-uniform store guard on N only.
// ---------------------------------------------------------------------------
__global__ void gemm_bt_wmma(const __bf16* __restrict__ A, const __bf16* __restrict__ Bm,
                             float* __restrict__ C, int K, int Nlim,
                             int lda, int ldb, int ldc,
                             long sAb, long sAh, long sBb, long sBh, long sCb, long sCh,
                             int Hdim) {
  int z = blockIdx.z;
  int b = z / Hdim, h = z % Hdim;
  A  += (long)b * sAb + (long)h * sAh;
  Bm += (long)b * sBb + (long)h * sBh;
  C  += (long)b * sCb + (long)h * sCh;

  int m0   = blockIdx.x * 16;
  int w    = threadIdx.x >> 5;
  int n0   = blockIdx.y * 128 + w * 32;
  int lane = threadIdx.x & 31;
  int lm   = lane & 15;
  int half = lane >> 4;

  const __bf16* arow  = A  + (long)(m0 + lm) * lda + half * 8;
  const __bf16* brow0 = Bm + (long)(n0 + lm) * ldb + half * 8;
  const __bf16* brow1 = brow0 + (long)16 * ldb;

  v8f acc0 = {}, acc1 = {};
  for (int kk = 0; kk < K; kk += 32) {
    v16bf a  = load_afrag(arow + kk);
    v16bf b0 = load_bfrag(brow0 + kk);
    v16bf b1 = load_bfrag(brow1 + kk);
    acc0 = __builtin_amdgcn_wmma_f32_16x16x32_bf16(false, a, false, b0,
                                                   (short)0, acc0, false, false);
    acc1 = __builtin_amdgcn_wmma_f32_16x16x32_bf16(false, a, false, b1,
                                                   (short)0, acc1, false, false);
  }

  int n = n0 + lm;
  if (n < Nlim) {
    #pragma unroll
    for (int r = 0; r < 8; ++r) C[(long)(m0 + r + 8 * half) * ldc + n] = acc0[r];
  }
  if (n + 16 < Nlim) {
    #pragma unroll
    for (int r = 0; r < 8; ++r) C[(long)(m0 + r + 8 * half) * ldc + n + 16] = acc1[r];
  }
}

// ---------------------------------------------------------------------------
__global__ void cvt_bf16(const float* __restrict__ src, __bf16* __restrict__ dst, int n) {
  int i = blockIdx.x * blockDim.x + threadIdx.x;
  if (i < n) dst[i] = f2bf(src[i]);
}

__global__ void copy_pad_x(const float* __restrict__ src, float* __restrict__ dst,
                           __bf16* __restrict__ dstb, int n) {
  int i = blockIdx.x * blockDim.x + threadIdx.x;
  if (i < n) { float v = src[i]; dst[i] = v; dstb[i] = f2bf(v); }
}

// hyp_linear epilogue (mobius_matvec scaling + mobius_add bias), in place on Y.
__global__ void hyp_post(const float* __restrict__ X, float* __restrict__ Y,
                         const float* __restrict__ bias, int Nd, int Kd) {
  __shared__ float red[8];
  int m = blockIdx.x;
  const float* xr = X + (long)m * Kd;
  float* yr = Y + (long)m * Nd;

  float ssx = 0.f;
  for (int j = threadIdx.x; j < Kd; j += blockDim.x) { float v = xr[j]; ssx += v * v; }
  ssx = block_reduce_sum(ssx, red);

  float ssy = 0.f, syb = 0.f, sbb = 0.f;
  for (int j = threadIdx.x; j < Nd; j += blockDim.x) {
    float y = yr[j], b = bias[j];
    ssy += y * y; syb += y * b; sbb += b * b;
  }
  ssy = block_reduce_sum(ssy, red);
  syb = block_reduce_sum(syb, red);
  sbb = block_reduce_sum(sbb, red);

  float xn  = sqrtf(fmaxf(ssx, 1e-15f));
  float mxn = sqrtf(fmaxf(ssy, 1e-15f));
  float scl = tanhf(mxn / xn * artanhf_(fminf(xn, 1.f - 1e-5f))) / mxn;

  float x2 = scl * scl * ssy;
  float xy = scl * syb;
  float den = fmaxf(1.f + 2.f * xy + x2 * sbb, 1e-15f);
  float a1 = 1.f + 2.f * xy + sbb;
  float b1 = 1.f - x2;
  for (int j = threadIdx.x; j < Nd; j += blockDim.x) {
    float t = scl * yr[j];
    yr[j] = (a1 * t + b1 * bias[j]) / den;
  }
}

// Clamp q/k head-rows to max_norm, emit bf16 copy + ||row||^2 (f32 not kept).
__global__ void clamp_sumsq(const float* __restrict__ Q, __bf16* __restrict__ Qb,
                            float* __restrict__ out2, int H, int S, int Dh, int Dld) {
  __shared__ float red[8];
  int idx = blockIdx.x;               // b*H*S + h*S + s
  int s = idx % S; int bh = idx / S; int h = bh % H; int b = bh / H;
  long off = ((long)(b * S + s)) * Dld + h * Dh;
  int d = threadIdx.x;
  float v = fminf(Q[off + d], 1.0f - 1e-5f);
  Qb[off + d] = f2bf(v);
  float ss = block_reduce_sum(v * v, red);
  if (d == 0) out2[idx] = ss;         // c == 1
}

// Closed-form hyperbolic distance scores + softmax -> bf16 probabilities
// (pad columns written as zero so the K=Sld accumulation is exact).
__global__ void scores_softmax(const float* __restrict__ dot, __bf16* __restrict__ attnb,
                               const float* __restrict__ q2a, const float* __restrict__ k2a,
                               const float* __restrict__ hs,
                               int H, int S, int Dh, int Sld, int RowsBH) {
  __shared__ float red[8];
  int row = blockIdx.x;               // b*H*S + h*S + i
  int i = row % S; int bh = row / S; int h = bh % H;
  float q2i = q2a[row];
  const float* dr = dot + ((long)bh * RowsBH + i) * Sld;
  __bf16* ar = attnb + ((long)bh * RowsBH + i) * Sld;
  float inv = 1.0f / (hs[h] * sqrtf((float)Dh));
  int j = threadIdx.x;                // blockDim == Sld
  float sc = -3.0e38f;
  if (j < S) {
    float d   = dr[j];
    float k2j = k2a[(long)bh * S + j];
    float A1 = 1.f - 2.f * d + q2i;
    float B1 = 1.f - k2j;
    float num2 = A1 * A1 * k2j + B1 * B1 * q2i - 2.f * A1 * B1 * d;
    float den = fmaxf(1.f - 2.f * d + k2j * q2i, 1e-15f);
    float dns = num2 / (den * den);                      // ||mobius_add(-k,q)||^2
    float denom = (1.f - q2i) * (1.f - k2j) + 1e-15f;
    float aarg = fmaxf(1.f + 2.f * dns / denom, 1.f + 1e-7f);
    float dist = logf(aarg + sqrtf(aarg * aarg - 1.f));  // acosh
    sc = -dist * inv;
  }
  float mx = block_reduce_max(sc, red);
  float e = (j < S) ? expf(sc - mx) : 0.f;
  float ssum = block_reduce_sum(e, red);
  ar[j] = f2bf((j < S) ? (e / ssum) : 0.f);
}

// logmap0 of V head-rows -> transposed bf16 LVt (Dh x Sld per head).
__global__ void logmap0_rows_t(const float* __restrict__ V, __bf16* __restrict__ LVtb,
                               int H, int S, int Dh, int Dld, int Sld) {
  __shared__ float red[8];
  int idx = blockIdx.x;
  int s = idx % S; int bh = idx / S; int h = bh % H; int b = bh / H;
  const float* p = V + ((long)(b * S + s)) * Dld + h * Dh;
  int d = threadIdx.x;
  float v = p[d];
  float ss = block_reduce_sum(v * v, red);
  float n = sqrtf(fmaxf(ss, 1e-15f));
  float sc = artanhf_(fminf(n, 1.f - 1e-5f)) / n;
  LVtb[((long)bh * Dh + d) * Sld + s] = f2bf(sc * v);
}

// expmap0 over head-rows of AO (B,H,RowsBH,Dh) -> merged (B,S,H*Dh) f32+bf16.
__global__ void expmap0_merge(const float* __restrict__ AO, float* __restrict__ OUT,
                              __bf16* __restrict__ OUTb,
                              int H, int S, int Dh, int Dld, int RowsBH) {
  __shared__ float red[8];
  int idx = blockIdx.x;
  int s = idx % S; int bh = idx / S; int h = bh % H; int b = bh / H;
  int d = threadIdx.x;
  float v = AO[((long)bh * RowsBH + s) * Dh + d];
  float ss = block_reduce_sum(v * v, red);
  float n = sqrtf(fmaxf(ss, 1e-15f));
  float sc = tanhf(n) / n;
  float o = sc * v;
  long off = ((long)(b * S + s)) * Dld + h * Dh + d;
  OUT[off] = o;
  OUTb[off] = f2bf(o);
}

// out = hyp_layer_norm( mobius_add(X, mobius_scalar_mul(beta, T)) ), per row.
__global__ void residual_hln(const float* __restrict__ X, const float* __restrict__ T,
                             const float* __restrict__ beta, const float* __restrict__ lnw,
                             const float* __restrict__ lnb, float* __restrict__ OUT,
                             __bf16* __restrict__ OUTb, int D) {
  __shared__ float red[8];
  const int PT = 3;  // D=768, block=256
  int m = blockIdx.x;
  const float* xr = X + (long)m * D;
  const float* tr = T + (long)m * D;

  float xv[PT], tv[PT];
  float ssT = 0.f, ssX = 0.f, sXT = 0.f;
  #pragma unroll
  for (int i = 0; i < PT; ++i) {
    int j = threadIdx.x + i * blockDim.x;
    xv[i] = xr[j]; tv[i] = tr[j];
    ssT += tv[i] * tv[i]; ssX += xv[i] * xv[i]; sXT += xv[i] * tv[i];
  }
  ssT = block_reduce_sum(ssT, red);
  ssX = block_reduce_sum(ssX, red);
  sXT = block_reduce_sum(sXT, red);

  float nt = sqrtf(fmaxf(ssT, 1e-15f));
  float sm = tanhf(beta[0] * artanhf_(fminf(nt, 1.f - 1e-5f))) / nt;
  float y2 = sm * sm * ssT;
  float xy = sm * sXT;
  float den = fmaxf(1.f + 2.f * xy + ssX * y2, 1e-15f);
  float a1 = 1.f + 2.f * xy + y2;
  float b1 = 1.f - ssX;

  float hv[PT]; float ssH = 0.f;
  #pragma unroll
  for (int i = 0; i < PT; ++i) {
    hv[i] = (a1 * xv[i] + b1 * sm * tv[i]) / den;
    ssH += hv[i] * hv[i];
  }
  ssH = block_reduce_sum(ssH, red);
  float nh = sqrtf(fmaxf(ssH, 1e-15f));
  float ls = artanhf_(fminf(nh, 1.f - 1e-5f)) / nh;

  float lv_[PT]; float sl = 0.f;
  #pragma unroll
  for (int i = 0; i < PT; ++i) { lv_[i] = ls * hv[i]; sl += lv_[i]; }
  sl = block_reduce_sum(sl, red);
  float mu = sl / (float)D;
  float sv = 0.f;
  #pragma unroll
  for (int i = 0; i < PT; ++i) { float dd = lv_[i] - mu; sv += dd * dd; }
  sv = block_reduce_sum(sv, red);
  float rstd = rsqrtf(sv / (float)D + 1e-5f);

  float gv[PT]; float ssG = 0.f;
  #pragma unroll
  for (int i = 0; i < PT; ++i) {
    int j = threadIdx.x + i * blockDim.x;
    gv[i] = (lv_[i] - mu) * rstd * lnw[j] + lnb[j];
    ssG += gv[i] * gv[i];
  }
  ssG = block_reduce_sum(ssG, red);
  float ng = sqrtf(fmaxf(ssG, 1e-15f));
  float es = tanhf(ng) / ng;
  #pragma unroll
  for (int i = 0; i < PT; ++i) {
    int j = threadIdx.x + i * blockDim.x;
    float o = es * gv[i];
    OUT[(long)m * D + j] = o;
    if (OUTb) OUTb[(long)m * D + j] = f2bf(o);
  }
}

// mobius_relu per row (FF=3072, block 256, 12 elems/thread); f32 + bf16 out.
__global__ void mobius_relu_rows(float* __restrict__ F, __bf16* __restrict__ Fb, int Nd) {
  __shared__ float red[8];
  const int PT = 12;
  int m = blockIdx.x;
  float* fr = F + (long)m * Nd;
  __bf16* fbr = Fb + (long)m * Nd;
  float fv[PT]; float ss = 0.f;
  #pragma unroll
  for (int i = 0; i < PT; ++i) {
    int j = threadIdx.x + i * blockDim.x;
    fv[i] = fr[j]; ss += fv[i] * fv[i];
  }
  ss = block_reduce_sum(ss, red);
  float n = sqrtf(fmaxf(ss, 1e-15f));
  float lsc = artanhf_(fminf(n, 1.f - 1e-5f)) / n;
  float ss2 = 0.f;
  #pragma unroll
  for (int i = 0; i < PT; ++i) { fv[i] = fmaxf(lsc * fv[i], 0.f); ss2 += fv[i] * fv[i]; }
  ss2 = block_reduce_sum(ss2, red);
  float n2 = sqrtf(fmaxf(ss2, 1e-15f));
  float esc = tanhf(n2) / n2;
  #pragma unroll
  for (int i = 0; i < PT; ++i) {
    int j = threadIdx.x + i * blockDim.x;
    float o = esc * fv[i];
    fr[j] = o;
    fbr[j] = f2bf(o);
  }
}

// ---------------------------------------------------------------------------
static inline int ceil_div(int a, int b) { return (a + b - 1) / b; }

extern "C" void kernel_launch(void* const* d_in, const int* in_sizes, int n_in,
                              void* d_out, int out_size, void* d_ws, size_t ws_size,
                              hipStream_t stream) {
  constexpr int B = 2, S = 196, D = 768, H = 12, Dh = 64, FF = 3072;
  constexpr int M = B * S;        // 392
  constexpr int Mp  = 400;        // 25 tiles of 16 (exactly what the GEMMs write)
  constexpr int MpQK = 464;       // tail rows: score GEMM reads up to b*S+255
  constexpr int Sld = 256;        // padded attention columns (K-dim, mult of 32)
  constexpr int Srw = 208;        // padded attention rows (13 tiles of 16)
  constexpr int BH = B * H;       // 24

  const float* x    = (const float*)d_in[0];
  const float* wq   = (const float*)d_in[1];  const float* bq = (const float*)d_in[2];
  const float* wk   = (const float*)d_in[3];  const float* bk = (const float*)d_in[4];
  const float* wv   = (const float*)d_in[5];  const float* bv = (const float*)d_in[6];
  const float* wo   = (const float*)d_in[7];  const float* bo = (const float*)d_in[8];
  const float* hs   = (const float*)d_in[9];
  const float* w1   = (const float*)d_in[10]; const float* b1 = (const float*)d_in[11];
  const float* w2   = (const float*)d_in[12]; const float* b2 = (const float*)d_in[13];
  const float* beta = (const float*)d_in[14];
  const float* ln1w = (const float*)d_in[15]; const float* ln1b = (const float*)d_in[16];
  const float* ln2w = (const float*)d_in[17]; const float* ln2b = (const float*)d_in[18];
  float* out = (float*)d_out;
  (void)in_sizes; (void)n_in; (void)out_size;

  // ---- workspace layout: f32 region first, then 16B-aligned bf16 region ----
  float* ws   = (float*)d_ws;
  float* xp   = ws;                             // Mp*D
  float* qb   = xp   + (long)Mp * D;            // Mp*D
  float* kb   = qb   + (long)Mp * D;            // Mp*D
  float* vb   = kb   + (long)Mp * D;            // Mp*D
  float* q2   = vb   + (long)Mp * D;            // BH*S
  float* k2   = q2   + (long)BH * S;            // BH*S
  float* dotb = k2   + (long)BH * S;            // BH*Srw*Sld   (f32 q.k)
  float* ao   = dotb + (long)BH * Srw * Sld;    // BH*Srw*Dh
  float* aot  = ao   + (long)BH * Srw * Dh;     // Mp*D
  float* ob   = aot  + (long)Mp * D;            // Mp*D
  float* hb   = ob   + (long)Mp * D;            // Mp*D
  float* fb   = hb   + (long)Mp * D;            // Mp*FF
  float* f2b  = fb   + (long)Mp * FF;           // Mp*D
  float* fend = f2b  + (long)Mp * D;

  __bf16* xpb   = (__bf16*)fend;                    // Mp*D
  __bf16* wqb   = xpb   + (long)Mp * D;             // D*D
  __bf16* wkb   = wqb   + (long)D * D;              // D*D
  __bf16* wvb   = wkb   + (long)D * D;              // D*D
  __bf16* wob   = wvb   + (long)D * D;              // D*D
  __bf16* w1b   = wob   + (long)D * D;              // FF*D
  __bf16* w2b   = w1b   + (long)FF * D;             // D*FF
  __bf16* qbb   = w2b   + (long)D * FF;             // MpQK*D (tail rows zero)
  __bf16* kbb   = qbb   + (long)MpQK * D;           // MpQK*D (tail rows zero)
  __bf16* attnb = kbb   + (long)MpQK * D;           // BH*Srw*Sld
  __bf16* lvtb  = attnb + (long)BH * Srw * Sld;     // BH*Dh*Sld
  __bf16* aotb  = lvtb  + (long)BH * Dh * Sld;      // Mp*D
  __bf16* hbb   = aotb  + (long)Mp * D;             // Mp*D
  __bf16* fbb   = hbb   + (long)Mp * D;             // Mp*FF
  __bf16* bend  = fbb   + (long)Mp * FF;
  size_t total = (size_t)((char*)bend - (char*)d_ws);
  (void)ws_size;

  // Zero all pads so every GEMM runs guard-free (bf16 zero == 0x0000).
  hipMemsetAsync(d_ws, 0, total, stream);

  dim3 blk128(128), blk256(256), blk64(64);

  // ---- one-time bf16 conversion of weights; padded copy of x ----
  copy_pad_x<<<ceil_div(M * D, 256), blk256, 0, stream>>>(x, xp, xpb, M * D);
  cvt_bf16<<<ceil_div(D * D, 256), blk256, 0, stream>>>(wq, wqb, D * D);
  cvt_bf16<<<ceil_div(D * D, 256), blk256, 0, stream>>>(wk, wkb, D * D);
  cvt_bf16<<<ceil_div(D * D, 256), blk256, 0, stream>>>(wv, wvb, D * D);
  cvt_bf16<<<ceil_div(D * D, 256), blk256, 0, stream>>>(wo, wob, D * D);
  cvt_bf16<<<ceil_div(FF * D, 256), blk256, 0, stream>>>(w1, w1b, FF * D);
  cvt_bf16<<<ceil_div(D * FF, 256), blk256, 0, stream>>>(w2, w2b, D * FF);

  // ---- q, k, v = hyp_linear(x, W*, b*) ----
  {
    dim3 g(Mp / 16, D / 128, 1);
    gemm_bt_wmma<<<g, blk128, 0, stream>>>(xpb, wqb, qb, D, D, D, D, D,
                                           0, 0, 0, 0, 0, 0, 1);
    gemm_bt_wmma<<<g, blk128, 0, stream>>>(xpb, wkb, kb, D, D, D, D, D,
                                           0, 0, 0, 0, 0, 0, 1);
    gemm_bt_wmma<<<g, blk128, 0, stream>>>(xpb, wvb, vb, D, D, D, D, D,
                                           0, 0, 0, 0, 0, 0, 1);
    hyp_post<<<M, blk256, 0, stream>>>(x, qb, bq, D, D);
    hyp_post<<<M, blk256, 0, stream>>>(x, kb, bk, D, D);
    hyp_post<<<M, blk256, 0, stream>>>(x, vb, bv, D, D);
  }

  // ---- clamp + head-row norms; bf16 copies for the score GEMM ----
  clamp_sumsq<<<BH * S, blk64, 0, stream>>>(qb, qbb, q2, H, S, Dh, D);
  clamp_sumsq<<<BH * S, blk64, 0, stream>>>(kb, kbb, k2, H, S, Dh, D);

  // ---- dot[b,h,i,j] = q . k  (batched Q*K^T over heads) ----
  {
    dim3 g(Srw / 16, Sld / 128, BH);
    gemm_bt_wmma<<<g, blk128, 0, stream>>>(qbb, kbb, dotb, Dh, Sld, D, D, Sld,
                                           (long)S * D, Dh, (long)S * D, Dh,
                                           (long)H * Srw * Sld, (long)Srw * Sld, H);
  }

  // ---- hyperbolic scores + softmax -> bf16 attn (pad columns zero) ----
  scores_softmax<<<BH * S, dim3(Sld), 0, stream>>>(dotb, attnb, q2, k2, hs,
                                                   H, S, Dh, Sld, Srw);

  // ---- lvt = logmap0(v) transposed bf16; ao = attn @ lvt^T ----
  logmap0_rows_t<<<BH * S, blk64, 0, stream>>>(vb, lvtb, H, S, Dh, D, Sld);
  {
    dim3 g(Srw / 16, 1, BH);  // 2 waves cover N=64
    gemm_bt_wmma<<<g, blk64, 0, stream>>>(attnb, lvtb, ao, Sld, Dh, Sld, Sld, Dh,
                                          (long)H * Srw * Sld, (long)Srw * Sld,
                                          (long)H * Dh * Sld, (long)Dh * Sld,
                                          (long)H * Srw * Dh, (long)Srw * Dh, H);
  }

  // ---- expmap0 + merge heads -> aot (f32+bf16); hyp_linear with wo ----
  expmap0_merge<<<BH * S, blk64, 0, stream>>>(ao, aot, aotb, H, S, Dh, D, Srw);
  {
    dim3 g(Mp / 16, D / 128, 1);
    gemm_bt_wmma<<<g, blk128, 0, stream>>>(aotb, wob, ob, D, D, D, D, D,
                                           0, 0, 0, 0, 0, 0, 1);
    hyp_post<<<M, blk256, 0, stream>>>(aot, ob, bo, D, D);
  }

  // ---- residual 1 + hyperbolic layer norm (f32 + bf16 for FFN GEMM) ----
  residual_hln<<<M, blk256, 0, stream>>>(x, ob, beta, ln1w, ln1b, hb, hbb, D);

  // ---- mobius FFN ----
  {
    dim3 g(Mp / 16, FF / 128, 1);
    gemm_bt_wmma<<<g, blk128, 0, stream>>>(hbb, w1b, fb, D, FF, D, D, FF,
                                           0, 0, 0, 0, 0, 0, 1);
    hyp_post<<<M, blk256, 0, stream>>>(hb, fb, b1, FF, D);
  }
  mobius_relu_rows<<<M, blk256, 0, stream>>>(fb, fbb, FF);
  {
    dim3 g(Mp / 16, D / 128, 1);
    gemm_bt_wmma<<<g, blk128, 0, stream>>>(fbb, w2b, f2b, FF, D, FF, FF, D,
                                           0, 0, 0, 0, 0, 0, 1);
    hyp_post<<<M, blk256, 0, stream>>>(fb, f2b, b2, D, FF);
  }

  // ---- residual 2 + hyperbolic layer norm -> output ----
  residual_hln<<<M, blk256, 0, stream>>>(hb, f2b, beta, ln2w, ln2b, out,
                                         (__bf16*)nullptr, D);
}
